// MultiHeadAttention_22067541966943
// MI455X (gfx1250) — matrix operand order repaired and checked by reading, not compiled, not run on hardware
//
#include <hip/hip_runtime.h>
#include <hip/hip_bf16.h>
#include <stdint.h>

// ---------------- problem constants ----------------
#define B_  4
#define S_  2048
#define D_  1024
#define H_  16
#define HD_ 64
#define M_  (B_ * S_)   // 8192 rows for all projection GEMMs

// 1/sqrt(HD) * log2(e): folded into the Q projection so attention runs in
// the log2 domain and every exp is a single v_exp_f32.
#define QSCALE 0.180336880111120419f

typedef __attribute__((ext_vector_type(16))) __bf16 v16bf;
typedef __attribute__((ext_vector_type(2)))  __bf16 v2bf;
typedef __attribute__((ext_vector_type(8)))  float  v8f;

// fp32 -> bf16, round-to-nearest-even (scalar path)
__device__ __forceinline__ __bf16 f2bf(float f) {
  uint32_t u = __builtin_bit_cast(uint32_t, f);
  uint32_t r = u + 0x7FFFu + ((u >> 16) & 1u);
  uint16_t h = (uint16_t)(r >> 16);
  return __builtin_bit_cast(__bf16, h);
}
__device__ __forceinline__ uint16_t bf_bits(__bf16 v) {
  return __builtin_bit_cast(uint16_t, v);
}
__device__ __forceinline__ __bf16 us2bf(uint16_t v) {
  return __builtin_bit_cast(__bf16, v);
}
// pack two f32 -> one dword of 2x bf16 (hardware v_cvt_pk_bf16_f32 if present)
__device__ __forceinline__ uint32_t pack2bf(float lo, float hi) {
#if __has_builtin(__builtin_amdgcn_cvt_pk_bf16_f32)
  v2bf t = __builtin_amdgcn_cvt_pk_bf16_f32(lo, hi);
  return __builtin_bit_cast(uint32_t, t);
#else
  return (uint32_t)bf_bits(f2bf(lo)) | ((uint32_t)bf_bits(f2bf(hi)) << 16);
#endif
}

// ---------------- bulk f32 -> bf16 conversion (one pass, 8 elems/thread) ----
__global__ void __launch_bounds__(256)
cvt_f32_to_bf16(const float* __restrict__ s, uint16_t* __restrict__ d, int n) {
  const int i = (blockIdx.x * 256 + threadIdx.x) * 8;
  if (i >= n) return;
  const float4 x = *(const float4*)(s + i);
  const float4 y = *(const float4*)(s + i + 4);
  *(uint4*)(d + i) = make_uint4(pack2bf(x.x, x.y), pack2bf(x.z, x.w),
                                pack2bf(y.x, y.y), pack2bf(y.z, y.w));
}

// ---------------- WMMA fragment loaders (bf16 sources) ----------------------
// A (16x32 bf16, MxK): lane l16 = row M; half=0 -> elems K0..15,
//                                       half=1 -> elems K16..31
__device__ __forceinline__ v16bf load_a_bf(const uint16_t* base, int ld,
                                           int row0, int k0, int l16, int half) {
  const uint16_t* p = base + (size_t)(row0 + l16) * ld + (k0 + half * 16);
  v16bf a;
#pragma unroll
  for (int j = 0; j < 16; ++j) a[j] = us2bf(p[j]);
  return a;
}
// B (32x16 bf16, KxN): lane l16 = col N; half=0 holds K0..15, half=1 K16..31.
// Source is B^T = [N][K] row-major (K contiguous), i.e. W[n][k] for x@W^T.
__device__ __forceinline__ v16bf load_b_bf(const uint16_t* base, int ld,
                                           int n0, int k0, int l16, int half) {
  const uint16_t* p = base + (size_t)(n0 + l16) * ld + (k0 + half * 16);
  v16bf b;
#pragma unroll
  for (int j = 0; j < 16; ++j) b[j] = us2bf(p[j]);
  return b;
}

__device__ __forceinline__ v8f wmma_bf16(v16bf a, v16bf b, v8f c) {
  return __builtin_amdgcn_wmma_f32_16x16x32_bf16(false, a, false, b,
                                                 (short)0, c, false, false);
}

// ---------------- GEMM: C[M,1024] = oscale * (A[M,1024] @ W[1024,1024]^T) ---
// 64x64 output tile per wave: 16 WMMAs per K=32 step, 8 fragment loads.
// OM: 0 = fp32 out, 1 = bf16 out, 2 = bf16 out transposed per head
//     ([b][h][hd][s]) for the V projection.
template <int OM>
__global__ void __launch_bounds__(256)
mha_gemm_wmma(const uint16_t* __restrict__ Aptr, const uint16_t* __restrict__ Wptr,
              void* __restrict__ Cptr, float oscale) {
  const int lane = threadIdx.x & 31;
  const int l16  = lane & 15;
  const int half = lane >> 4;
  const int gw   = blockIdx.x * 8 + (threadIdx.x >> 5);
  const int TN   = D_ / 64;                  // 16 column tiles
  const int tm   = gw / TN, tn = gw % TN;
  const int row0 = tm * 64, col0 = tn * 64;

  v8f acc[4][4] = {};
  for (int k0 = 0; k0 < D_; k0 += 32) {
    v16bf a[4], b[4];
#pragma unroll
    for (int i = 0; i < 4; ++i)
      a[i] = load_a_bf(Aptr, D_, row0 + 16 * i, k0, l16, half);
#pragma unroll
    for (int j = 0; j < 4; ++j)
      b[j] = load_b_bf(Wptr, D_, col0 + 16 * j, k0, l16, half);
    if (k0 + 64 < D_) {                      // prefetch next-next k-step lines
#pragma unroll
      for (int i = 0; i < 4; ++i)
        __builtin_prefetch(Aptr + (size_t)(row0 + 16 * i + l16) * D_ + k0 + 64, 0, 1);
#pragma unroll
      for (int j = 0; j < 4; ++j)
        __builtin_prefetch(Wptr + (size_t)(col0 + 16 * j + l16) * D_ + k0 + 64, 0, 1);
    }
#pragma unroll
    for (int i = 0; i < 4; ++i)
#pragma unroll
      for (int j = 0; j < 4; ++j)
        acc[i][j] = wmma_bf16(a[i], b[j], acc[i][j]);
  }
#pragma unroll
  for (int i = 0; i < 4; ++i)
#pragma unroll
    for (int jf = 0; jf < 4; ++jf)
#pragma unroll
      for (int r = 0; r < 8; ++r) {
        const int row = row0 + i * 16 + r + half * 8;   // C layout (ISA 7.12.2)
        const int col = col0 + jf * 16 + l16;
        const float v = acc[i][jf][r] * oscale;
        if (OM == 0) {
          ((float*)Cptr)[(size_t)row * D_ + col] = v;
        } else if (OM == 1) {
          ((uint16_t*)Cptr)[(size_t)row * D_ + col] = bf_bits(f2bf(v));
        } else {
          const int bb = row >> 11, ss = row & (S_ - 1);
          const int h  = col / HD_, hd = col % HD_;
          ((uint16_t*)Cptr)[(((size_t)bb * H_ + h) * HD_ + hd) * S_ + ss] =
              bf_bits(f2bf(v));
        }
      }
}

// ---------------- fused causal flash attention (per-head) -------------------
// One wave owns 32 query rows (2 row-frags) of one (b,h); streams 32-key tiles
// with online softmax in the log2 domain (Q pre-scaled by QSCALE).
// xq/xk: bf16 [b][s][h][hd]; xvT: bf16 [b][h][hd][s]; out: bf16 [b][s][h][hd].
__global__ void __launch_bounds__(256)
mha_flash_attn(const uint16_t* __restrict__ xq, const uint16_t* __restrict__ xk,
               const uint16_t* __restrict__ xvT, uint16_t* __restrict__ aout) {
  __shared__ uint16_t psh[8][32 * 32];        // per-wave P bounce buffer
  const int lane = threadIdx.x & 31;
  const int l16  = lane & 15;
  const int half = lane >> 4;
  const int wid  = threadIdx.x >> 5;
  const int gw   = blockIdx.x * 8 + wid;
  const int QT   = S_ / 32;                   // 64 query tiles per head
  const int qt   = gw % QT;
  const int h    = (gw / QT) % H_;
  const int b    = gw / (QT * H_);
  const int q0   = qt * 32;

  const uint16_t* qbase = xq + (size_t)b * S_ * D_ + h * HD_;      // ld = D_
  const uint16_t* kbase = xk + (size_t)b * S_ * D_ + h * HD_;      // ld = D_
  const uint16_t* vbase = xvT + ((size_t)b * H_ + h) * HD_ * S_;   // ld = S_

  v16bf aq[2][2];                             // [row frag][k step]
#pragma unroll
  for (int qr = 0; qr < 2; ++qr) {
    aq[qr][0] = load_a_bf(qbase, D_, q0 + qr * 16, 0,  l16, half);
    aq[qr][1] = load_a_bf(qbase, D_, q0 + qr * 16, 32, l16, half);
  }

  v8f   o[2][4] = {};
  float mrow[2][8], lrow[2][8];
#pragma unroll
  for (int qr = 0; qr < 2; ++qr)
#pragma unroll
    for (int r = 0; r < 8; ++r) { mrow[qr][r] = -__builtin_inff(); lrow[qr][r] = 0.f; }

  const int ktmax = (q0 + 31) / 32;
  for (int kt = 0; kt <= ktmax; ++kt) {
    const int kb0 = kt * 32;
    // K-tile B fragments: [col frag][k step], shared by both query row-frags
    v16bf bk[2][2];
#pragma unroll
    for (int cf = 0; cf < 2; ++cf) {
      bk[cf][0] = load_b_bf(kbase, D_, kb0 + cf * 16, 0,  l16, half);
      bk[cf][1] = load_b_bf(kbase, D_, kb0 + cf * 16, 32, l16, half);
    }
    if (kt < ktmax) {                         // prefetch next key tile K + V
#pragma unroll
      for (int cf = 0; cf < 2; ++cf)
        __builtin_prefetch(kbase + (size_t)(kb0 + 32 + cf * 16 + l16) * D_, 0, 1);
#pragma unroll
      for (int f = 0; f < 4; ++f)
        __builtin_prefetch(vbase + (size_t)(f * 16 + l16) * S_ + kb0 + 32, 0, 1);
    }

#pragma unroll
    for (int qr = 0; qr < 2; ++qr) {
      // S tile 16(q) x 32(k): 4 WMMAs (scores already in log2 domain)
      v8f sc[2] = {};
#pragma unroll
      for (int cf = 0; cf < 2; ++cf) {
        sc[cf] = wmma_bf16(aq[qr][0], bk[cf][0], sc[cf]);
        sc[cf] = wmma_bf16(aq[qr][1], bk[cf][1], sc[cf]);
      }
      // causal mask (data select, EXEC stays all-ones)
#pragma unroll
      for (int cf = 0; cf < 2; ++cf) {
        const int key = kb0 + cf * 16 + l16;
#pragma unroll
        for (int r = 0; r < 8; ++r) {
          const int qrow = q0 + qr * 16 + r + half * 8;
          sc[cf][r] = (key <= qrow) ? sc[cf][r] : -__builtin_inff();
        }
      }
      // online softmax; a row lives in 16 lanes of one half-wave
#pragma unroll
      for (int r = 0; r < 8; ++r) {
        float tmax = fmaxf(sc[0][r], sc[1][r]);
#pragma unroll
        for (int mm = 1; mm < 16; mm <<= 1) tmax = fmaxf(tmax, __shfl_xor(tmax, mm, 16));
        const float mnew  = fmaxf(mrow[qr][r], tmax);
        const float scale = exp2f(mrow[qr][r] - mnew);      // 1x v_exp_f32
        const float p0 = exp2f(sc[0][r] - mnew);
        const float p1 = exp2f(sc[1][r] - mnew);
        float ps = p0 + p1;
#pragma unroll
        for (int mm = 1; mm < 16; mm <<= 1) ps += __shfl_xor(ps, mm, 16);
        lrow[qr][r] = lrow[qr][r] * scale + ps;
        mrow[qr][r] = mnew;
#pragma unroll
        for (int f = 0; f < 4; ++f) o[qr][f][r] *= scale;
        // stage P (C layout -> LDS row-major 32x32). Same-wave LDS ops are
        // in-order (DScnt), so the reads below see this data w/o a barrier.
        const int prow = qr * 16 + r + half * 8;
        psh[wid][prow * 32 + l16]      = bf_bits(f2bf(p0));
        psh[wid][prow * 32 + 16 + l16] = bf_bits(f2bf(p1));
      }
    }

    // PV: V^T B-fragments loaded once, feed both query row-frags (8 WMMAs)
#pragma unroll
    for (int f = 0; f < 4; ++f) {
      const v16bf bv = load_b_bf(vbase, S_, f * 16, kb0, l16, half);
#pragma unroll
      for (int qr = 0; qr < 2; ++qr) {
        const v16bf pa = load_a_bf(&psh[wid][0], 32, qr * 16, 0, l16, half);
        o[qr][f] = wmma_bf16(pa, bv, o[qr][f]);
      }
    }
  }

  // normalize + store [b][s][h][hd] bf16
#pragma unroll
  for (int qr = 0; qr < 2; ++qr)
#pragma unroll
    for (int r = 0; r < 8; ++r) {
      const float rinv = 1.0f / lrow[qr][r];
      const int   row  = q0 + qr * 16 + r + half * 8;
#pragma unroll
      for (int f = 0; f < 4; ++f) {
        const int col = f * 16 + l16;
        aout[((size_t)(b * S_ + row) * H_ + h) * HD_ + col] =
            bf_bits(f2bf(o[qr][f][r] * rinv));
      }
    }
}

// ---------------- host-side orchestration ----------------------------------
extern "C" void kernel_launch(void* const* d_in, const int* in_sizes, int n_in,
                              void* d_out, int out_size, void* d_ws, size_t ws_size,
                              hipStream_t stream) {
  (void)in_sizes; (void)n_in; (void)out_size;
  const float* q  = (const float*)d_in[0];
  const float* k  = (const float*)d_in[1];
  const float* v  = (const float*)d_in[2];
  const float* wq = (const float*)d_in[3];
  const float* wk = (const float*)d_in[4];
  const float* wv = (const float*)d_in[5];
  const float* wo = (const float*)d_in[6];

  const size_t E  = (size_t)M_ * D_;          // 8,388,608 elems per activation
  const size_t WE = (size_t)D_ * D_;          // 1,048,576 elems per weight
  const size_t need = (6 * E + 4 * WE) * sizeof(uint16_t);   // ~104 MB
  if (ws_size < need) return;                 // workspace contract not met

  uint16_t* qbf  = (uint16_t*)d_ws;           // slot 0 (reused for attn out)
  uint16_t* kbf  = qbf + E;                   // slot 1
  uint16_t* vbf  = kbf + E;                   // slot 2
  uint16_t* xq   = vbf + E;                   // slot 3: bf16 [b][s][h][hd], pre-scaled
  uint16_t* xk   = xq + E;                    // slot 4: bf16 [b][s][h][hd]
  uint16_t* xvT  = xk + E;                    // slot 5: bf16 [b][h][hd][s]
  uint16_t* wqb  = xvT + E;
  uint16_t* wkb  = wqb + WE;
  uint16_t* wvb  = wkb + WE;
  uint16_t* wob  = wvb + WE;
  uint16_t* attn = qbf;                       // q/k/v bf16 dead by attention time

  const dim3 blk(256);
  const dim3 gemm_grid((M_ / 64) * (D_ / 64) / 8);   // 2048 waves / 8 per block
  const dim3 attn_grid(B_ * H_ * (S_ / 32) / 8);     // 4096 waves / 8 per block
  const dim3 cvtA_grid((int)(E / 2048));             // 8 elems/thread
  const dim3 cvtW_grid((int)(WE / 2048));

  cvt_f32_to_bf16<<<cvtA_grid, blk, 0, stream>>>(q, qbf, (int)E);
  cvt_f32_to_bf16<<<cvtA_grid, blk, 0, stream>>>(k, kbf, (int)E);
  cvt_f32_to_bf16<<<cvtA_grid, blk, 0, stream>>>(v, vbf, (int)E);
  cvt_f32_to_bf16<<<cvtW_grid, blk, 0, stream>>>(wq, wqb, (int)WE);
  cvt_f32_to_bf16<<<cvtW_grid, blk, 0, stream>>>(wk, wkb, (int)WE);
  cvt_f32_to_bf16<<<cvtW_grid, blk, 0, stream>>>(wv, wvb, (int)WE);
  cvt_f32_to_bf16<<<cvtW_grid, blk, 0, stream>>>(wo, wob, (int)WE);

  // Q projection pre-scaled by 1/sqrt(HD)*log2(e) -> attention exp == exp2
  mha_gemm_wmma<1><<<gemm_grid, blk, 0, stream>>>(qbf, wqb, xq, QSCALE);
  mha_gemm_wmma<1><<<gemm_grid, blk, 0, stream>>>(kbf, wkb, xk, 1.0f);
  mha_gemm_wmma<2><<<gemm_grid, blk, 0, stream>>>(vbf, wvb, xvT, 1.0f);
  mha_flash_attn<<<attn_grid, blk, 0, stream>>>(xq, xk, xvT, attn);
  mha_gemm_wmma<0><<<gemm_grid, blk, 0, stream>>>(attn, wob, (float*)d_out, 1.0f);
}
